// ClassConditionalAffinity_74311524155914
// MI455X (gfx1250) — matrix operand order repaired and checked by reading, not compiled, not run on hardware
//
#include <hip/hip_runtime.h>
#include <hip/hip_bf16.h>

typedef __attribute__((ext_vector_type(16))) _Float16 v16h;
typedef __attribute__((ext_vector_type(8)))  float    v8f;
typedef __attribute__((ext_vector_type(4)))  float    v4f;

// Fixed problem sizes from the reference
constexpr int B_     = 4;
constexpr int D_     = 256;
constexpr int Wdim   = 64;
constexpr int HW     = 64 * 64;      // 4096
constexpr int STRIDE = 4;            // H / grid_size
constexpr int G2     = 256;          // 16*16 grid points
constexpr int P_     = 960;          // directed 4-neighborhood pairs
constexpr int COORD  = 128;
constexpr int MLP_IN = 640;          // 2*D + COORD
constexpr int H1     = 256;
constexpr int H2     = 128;
constexpr int ROWS   = B_ * P_;      // 3840
constexpr int TILES  = ROWS / 16;    // 240
constexpr int KT1    = MLP_IN / 32;  // 20
constexpr int NT1    = H1 / 16;      // 16
constexpr int KT2    = H1 / 32;      // 8
constexpr int NT2    = H2 / 16;      // 8

// ---------------------------------------------------------------------------
// 1) Pair builder: reproduces np.nonzero order (sorted by p, then q).
// ---------------------------------------------------------------------------
__global__ void build_pairs_kernel(int* __restrict__ idx1, int* __restrict__ idx2,
                                   int* __restrict__ rowstart) {
  __shared__ int cnt[G2];
  __shared__ int off[G2 + 1];
  const int g = threadIdx.x;                  // 256 threads
  const int gi = (g >> 4) * STRIDE;
  const int gj = (g & 15) * STRIDE;

  int c = 0;
  for (int q = 0; q < G2; ++q) {
    if (q == g) continue;
    int qi = (q >> 4) * STRIDE, qj = (q & 15) * STRIDE;
    int di = gi - qi; if (di < 0) di = -di;
    int dj = gj - qj; if (dj < 0) dj = -dj;
    if (di + dj <= 5) ++c;
  }
  cnt[g] = c;
  __syncthreads();
  if (g == 0) {
    int s = 0;
    for (int i = 0; i < G2; ++i) { off[i] = s; s += cnt[i]; }
    off[G2] = s;
  }
  __syncthreads();
  rowstart[g] = off[g];
  if (g == 0) rowstart[G2] = off[G2];
  int w = off[g];
  for (int q = 0; q < G2; ++q) {
    if (q == g) continue;
    int qi = (q >> 4) * STRIDE, qj = (q & 15) * STRIDE;
    int di = gi - qi; if (di < 0) di = -di;
    int dj = gj - qj; if (dj < 0) dj = -dj;
    if (di + dj <= 5) {
      idx1[w] = gi * Wdim + gj;
      idx2[w] = qi * Wdim + qj;
      ++w;
    }
  }
}

// ---------------------------------------------------------------------------
// 2) Streaming identity fill: 268 MB of NT b128 stores (the roofline term).
// ---------------------------------------------------------------------------
__global__ void fill_identity_kernel(float* __restrict__ out) {
  const unsigned v = blockIdx.x * blockDim.x + threadIdx.x;  // float4 index
  const int row  = (int)((v >> 10) & (HW - 1));              // 1024 float4 per row
  const int col0 = (int)(v & 1023) << 2;
  v4f val = {0.f, 0.f, 0.f, 0.f};
  const int d = row - col0;
  if (d >= 0 && d < 4) val[d] = 1.0f;
  __builtin_nontemporal_store(val, ((v4f*)out) + v);
}

// ---------------------------------------------------------------------------
// 3) Weight packer: f32 (K x N) row-major -> f16 WMMA B-fragment layout:
//    out[((nt*KT + kt)*32 + lane)*16 + j] = W[kt*32 + grp*8 + (j&7) + (j>>3)*16][nt*16 + ln]
//    (grp = lane>>4, ln = lane&15).  Each lane's fragment = 32 contiguous B.
// ---------------------------------------------------------------------------
__global__ void pack_weights_kernel(const float* __restrict__ W,
                                    const int* __restrict__ cls_p,
                                    _Float16* __restrict__ out, int K, int N) {
  const int t = blockIdx.x * blockDim.x + threadIdx.x;
  if (t >= K * N) return;
  const int j    = t & 15;
  const int lane = (t >> 4) & 31;
  const int fi   = t >> 9;            // nt*KT + kt
  const int KT   = K >> 5;
  const int kt = fi % KT, nt = fi / KT;
  const int grp = lane >> 4, ln = lane & 15;
  const int k = kt * 32 + grp * 8 + (j & 7) + ((j >> 3) << 4);
  const int n = nt * 16 + ln;
  const float* Wc = W + (size_t)cls_p[0] * K * N;
  out[t] = (_Float16)Wc[(size_t)k * N + n];
}

// ---------------------------------------------------------------------------
// 4) WMMA MLP: one wave per 16-row tile. f16 operands, f32 accum.
// ---------------------------------------------------------------------------
__global__ void __launch_bounds__(32) mlp_pairs_kernel(
    const float* __restrict__ features, const int* __restrict__ cls_p,
    const float* __restrict__ emb,
    const v16h* __restrict__ w1p, const float* __restrict__ b1,
    const v16h* __restrict__ w2p, const float* __restrict__ b2,
    const float* __restrict__ W3, const float* __restrict__ b3,
    const int* __restrict__ idx1, const int* __restrict__ idx2,
    float* __restrict__ vals) {
  __shared__ _Float16 xs [16 * MLP_IN];   // 20 KB
  __shared__ _Float16 h1s[16 * H1];       //  8 KB
  __shared__ _Float16 h2s[16 * H2];       //  4 KB

  const int lane = threadIdx.x;
  const int tile = blockIdx.x;
  const int cls  = cls_p[0];
  const float* b1c = b1 + cls * H1;
  const float* b2c = b2 + cls * H2;
  const float* W3c = W3 + cls * H2;
  const float  b3c = b3[cls];

  // Stage x-tile (16 rows x 640) into LDS as f16.
  for (int t = lane; t < 16 * MLP_IN; t += 32) {
    const int m = t / MLP_IN, k = t - m * MLP_IN;
    const int r = tile * 16 + m;
    const int b = r / P_, p = r - b * P_;
    const int i1 = idx1[p], i2 = idx2[p];
    float x;
    if (k < D_) {
      x = features[(size_t)b * D_ * HW + (size_t)k * HW + i1];
    } else if (k < 2 * D_) {
      x = features[(size_t)b * D_ * HW + (size_t)(k - D_) * HW + i2];
    } else {
      const int kc = k - 2 * D_;
      x = 0.5f * (emb[(size_t)i1 * COORD + kc] + emb[(size_t)i2 * COORD + kc]);
    }
    xs[t] = (_Float16)x;
  }
  __syncthreads();

  const int grp = lane >> 4;   // K-group for A fragments
  const int ln  = lane & 15;   // M (A) / N (B,C)

  // ---- Layer 1: (16x640) @ (640x256), relu ----
  for (int nt = 0; nt < NT1; ++nt) {
    v8f acc = {0.f, 0.f, 0.f, 0.f, 0.f, 0.f, 0.f, 0.f};
#pragma unroll 4
    for (int kt = 0; kt < KT1; ++kt) {
      v16h a;
      const int kbA = kt * 32 + grp * 8;
#pragma unroll
      for (int j = 0; j < 8; ++j) a[j]     = xs[ln * MLP_IN + kbA + j];
#pragma unroll
      for (int j = 0; j < 8; ++j) a[8 + j] = xs[ln * MLP_IN + kbA + 16 + j];
      const v16h bfrag = w1p[(nt * KT1 + kt) * 32 + lane];  // 32B coalesced
      acc = __builtin_amdgcn_wmma_f32_16x16x32_f16(false, a, false, bfrag,
                                                   (short)0, acc, false, false);
    }
    const int n = nt * 16 + ln;
    const float bias = b1c[n];
#pragma unroll
    for (int i = 0; i < 8; ++i) {
      float hv = acc[i] + bias;
      hv = hv > 0.f ? hv : 0.f;
      h1s[(grp * 8 + i) * H1 + n] = (_Float16)hv;
    }
  }
  __syncthreads();

  // ---- Layer 2: (16x256) @ (256x128), relu ----
  for (int nt = 0; nt < NT2; ++nt) {
    v8f acc = {0.f, 0.f, 0.f, 0.f, 0.f, 0.f, 0.f, 0.f};
#pragma unroll
    for (int kt = 0; kt < KT2; ++kt) {
      v16h a;
      const int kbA = kt * 32 + grp * 8;
#pragma unroll
      for (int j = 0; j < 8; ++j) a[j]     = h1s[ln * H1 + kbA + j];
#pragma unroll
      for (int j = 0; j < 8; ++j) a[8 + j] = h1s[ln * H1 + kbA + 16 + j];
      const v16h bfrag = w2p[(nt * KT2 + kt) * 32 + lane];
      acc = __builtin_amdgcn_wmma_f32_16x16x32_f16(false, a, false, bfrag,
                                                   (short)0, acc, false, false);
    }
    const int n = nt * 16 + ln;
    const float bias = b2c[n];
#pragma unroll
    for (int i = 0; i < 8; ++i) {
      float hv = acc[i] + bias;
      hv = hv > 0.f ? hv : 0.f;
      h2s[(grp * 8 + i) * H2 + n] = (_Float16)hv;
    }
  }
  __syncthreads();

  // ---- Layer 3: (16x128) @ (128x1), sigmoid ----
  if (lane < 16) {
    float acc = b3c;
    for (int k = 0; k < H2; ++k)
      acc += (float)h2s[lane * H2 + k] * W3c[k];
    vals[tile * 16 + lane] = 1.0f / (1.0f + __expf(-acc));
  }
}

// ---------------------------------------------------------------------------
// 5) Row-normalize + scatter vals into A.
// ---------------------------------------------------------------------------
__global__ void scatter_normalize_kernel(const float* __restrict__ vals,
                                         const int* __restrict__ idx2,
                                         const int* __restrict__ rowstart,
                                         float* __restrict__ out) {
  const int t = blockIdx.x * blockDim.x + threadIdx.x;
  if (t >= B_ * G2) return;
  const int b = t >> 8, g = t & 255;
  const int row = ((g >> 4) * STRIDE) * Wdim + (g & 15) * STRIDE;
  const int ps = rowstart[g], pe = rowstart[g + 1];
  float sum = 1.0f;
  for (int p = ps; p < pe; ++p) sum += vals[b * P_ + p];
  const float inv = 1.0f / (sum + 1e-8f);
  const size_t base = (size_t)b * HW * HW + (size_t)row * HW;
  out[base + row] = inv;
  for (int p = ps; p < pe; ++p)
    out[base + idx2[p]] = vals[b * P_ + p] * inv;
}

// ---------------------------------------------------------------------------
extern "C" void kernel_launch(void* const* d_in, const int* in_sizes, int n_in,
                              void* d_out, int out_size, void* d_ws, size_t ws_size,
                              hipStream_t stream) {
  const float* features = (const float*)d_in[0];
  const int*   cls      = (const int*)d_in[1];
  // d_in[2]=H, d_in[3]=W, d_in[4]=grid_size: fixed by the reference, hardcoded.
  const float* emb = (const float*)d_in[5];
  const float* W1  = (const float*)d_in[6];
  const float* b1  = (const float*)d_in[7];
  const float* W2  = (const float*)d_in[8];
  const float* b2  = (const float*)d_in[9];
  const float* W3  = (const float*)d_in[10];
  const float* b3  = (const float*)d_in[11];
  float* out = (float*)d_out;

  // Workspace layout (offsets keep w1p 32B-aligned given 256B-aligned d_ws):
  int*       idx1     = (int*)d_ws;                       // 960 ints
  int*       idx2     = idx1 + P_;                        // 960 ints
  int*       rowstart = idx2 + P_;                        // 257 ints (pad 272)
  float*     vals     = (float*)(rowstart + 272);         // 3840 floats
  _Float16*  w1p      = (_Float16*)(vals + ROWS);         // 640*256 halves
  _Float16*  w2p      = w1p + MLP_IN * H1;                // 256*128 halves

  build_pairs_kernel<<<1, G2, 0, stream>>>(idx1, idx2, rowstart);

  pack_weights_kernel<<<(MLP_IN * H1 + 255) / 256, 256, 0, stream>>>(
      W1, cls, w1p, MLP_IN, H1);
  pack_weights_kernel<<<(H1 * H2 + 255) / 256, 256, 0, stream>>>(
      W2, cls, w2p, H1, H2);

  const int fill_total = B_ * HW * (HW / 4);   // float4 stores
  fill_identity_kernel<<<fill_total / 256, 256, 0, stream>>>(out);

  mlp_pairs_kernel<<<TILES, 32, 0, stream>>>(features, cls, emb,
                                             (const v16h*)w1p, b1,
                                             (const v16h*)w2p, b2,
                                             W3, b3, idx1, idx2, vals);

  scatter_normalize_kernel<<<(B_ * G2 + 255) / 256, 256, 0, stream>>>(
      vals, idx2, rowstart, out);
}
